// RAFTStereo_43026982371456
// MI455X (gfx1250) — matrix operand order repaired and checked by reading, not compiled, not run on hardware
//
#include <hip/hip_runtime.h>
#include <hip/hip_bf16.h>

// Problem constants (match reference)
#define Bn   8
#define Cc   256
#define Hh   80
#define Wl   160
#define DISP 24
#define FAC  8
#define WT   (Wl / 16)      // 10 column tiles of 16
#define HW   (Hh * Wl)

typedef __attribute__((ext_vector_type(16))) _Float16 v16h;
typedef __attribute__((ext_vector_type(8)))  float    v8f;

// ---------------------------------------------------------------------------
// Stage 1: banded-GEMM cost volume via WMMA f16 16x16x32.
// One wave (32 threads) per (b, h, wtile). Computes 3 diagonal 16x16 tiles of
// L^T R covering the band 0 <= w - w' < 24; K loop over C=256 fully unrolled
// in chunks of 32. All loads are unconditional (clamped addresses + 0/1 lane
// mask) so they can be clause-batched; no divergent branches in the hot loop.
// ---------------------------------------------------------------------------
__global__ __launch_bounds__(32)
void cost_volume_wmma(const float* __restrict__ L,
                      const float* __restrict__ R,
                      float* __restrict__ cost)
{
    const int blk  = blockIdx.x;          // b*H*WT + h*WT + wt
    const int wt   = blk % WT;
    const int bh   = blk / WT;
    const int h    = bh % Hh;
    const int b    = bh / Hh;
    const int lane = threadIdx.x & 31;    // wave32
    const int half = lane >> 4;           // 0: lanes 0-15, 1: lanes 16-31
    const int ln   = lane & 15;

    const float* __restrict__ Lbh = L + (size_t)b * Cc * HW + (size_t)h * Wl;
    const float* __restrict__ Rbh = R + (size_t)b * Cc * HW + (size_t)h * Wl;

    // Per-lane A pointer: column w = wt*16 + ln; rows (c) advance by HW.
    const float* __restrict__ Ap = Lbh + wt * 16 + ln;

    // Per-lane B pointers for the 3 diagonal tiles. Negative w' is clamped to
    // a safe in-bounds address and the loaded value is zeroed branchlessly.
    const float* Bp[3];
    float bmask[3];
    #pragma unroll
    for (int d = 0; d < 3; ++d) {
        const int wp  = (wt - d) * 16 + ln;
        const int wps = wp < 0 ? 0 : wp;
        Bp[d]    = Rbh + wps;
        bmask[d] = wp < 0 ? 0.0f : 1.0f;
    }

    v8f acc[3];
    acc[0] = v8f{}; acc[1] = v8f{}; acc[2] = v8f{};

    #pragma unroll
    for (int kc = 0; kc < Cc; kc += 32) {
        // A tile -> 16-bit A-matrix 16x32 layout:
        // lane (half,ln): M = ln, element j holds K = ((j>>3)<<4)+(half<<3)+(j&7)
        v16h a;
        #pragma unroll
        for (int j = 0; j < 16; ++j) {
            const int c = kc + ((j >> 3) << 4) + (half << 3) + (j & 7);
            a[j] = (_Float16)Ap[(size_t)c * HW];
        }

        // B tiles -> 32x16 16-bit layout: N = ln, element j holds K = ((j>>3)<<4)+(half<<3)+(j&7)
        #pragma unroll
        for (int d = 0; d < 3; ++d) {
            v16h bm;
            #pragma unroll
            for (int j = 0; j < 16; ++j) {
                const int c = kc + ((j >> 3) << 4) + (half << 3) + (j & 7);
                bm[j] = (_Float16)(Bp[d][(size_t)c * HW] * bmask[d]);
            }
            acc[d] = __builtin_amdgcn_wmma_f32_16x16x32_f16(
                /*neg_a=*/false, a, /*neg_b=*/false, bm,
                /*c_mod=*/(short)0, acc[d],
                /*reuse_a=*/false, /*reuse_b=*/false);
        }
    }

    // Scatter band entries: C/D layout VGPR r -> M = r + 8*half, N = ln.
    const float scale = 1.0f / (float)Cc;
    #pragma unroll
    for (int d = 0; d < 3; ++d) {
        #pragma unroll
        for (int r = 0; r < 8; ++r) {
            const int m  = r + (half << 3);
            const int w  = wt * 16 + m;
            const int wp = (wt - d) * 16 + ln;
            const int i  = w - wp;                 // disparity index
            if (i >= 0 && i < DISP) {
                cost[(((size_t)b * DISP + i) * Hh + h) * Wl + w] = acc[d][r] * scale;
            }
        }
    }
}

// ---------------------------------------------------------------------------
// Stage 2: top-3 over disparity axis, softmax-weighted index expectation.
// One thread per (b,h,w) pixel; 24 strided reads, coalesced along w.
// ---------------------------------------------------------------------------
__global__ __launch_bounds__(256)
void topk_regress(const float* __restrict__ cost, float* __restrict__ disp)
{
    const int idx = blockIdx.x * blockDim.x + threadIdx.x;
    if (idx >= Bn * HW) return;
    const int w = idx % Wl;
    const int h = (idx / Wl) % Hh;
    const int b = idx / HW;

    const float* __restrict__ c = cost + (size_t)b * DISP * HW + (size_t)h * Wl + w;

    float v0 = -3.4e38f, v1 = -3.4e38f, v2 = -3.4e38f;
    int   i0 = 0, i1 = 0, i2 = 0;
    #pragma unroll
    for (int i = 0; i < DISP; ++i) {
        const float v = c[(size_t)i * HW];
        if (v > v0)      { v2 = v1; i2 = i1; v1 = v0; i1 = i0; v0 = v; i0 = i; }
        else if (v > v1) { v2 = v1; i2 = i1; v1 = v;  i1 = i; }
        else if (v > v2) { v2 = v;  i2 = i; }
    }
    const float e1 = __expf(v1 - v0);
    const float e2 = __expf(v2 - v0);
    const float s  = 1.0f + e1 + e2;
    disp[idx] = ((float)i0 + (float)i1 * e1 + (float)i2 * e2) / s;
}

// ---------------------------------------------------------------------------
// Stage 3: convex upsample (softmax over 9 neighbors, temp from iters) and
// negate. One thread per output pixel of [B, 1, 8H, 8W]. Bandwidth-bound on
// the 236 MB mask tensor.
// ---------------------------------------------------------------------------
__global__ __launch_bounds__(256)
void convex_upsample_neg(const float* __restrict__ disp,
                         const float* __restrict__ mask,
                         const int*   __restrict__ itersPtr,
                         float* __restrict__ out)
{
    const int total = Bn * (FAC * Hh) * (FAC * Wl);
    const int idx = blockIdx.x * blockDim.x + threadIdx.x;
    if (idx >= total) return;

    const int X = idx % (FAC * Wl);
    const int Y = (idx / (FAC * Wl)) % (FAC * Hh);
    const int b = idx / ((FAC * Wl) * (FAC * Hh));
    const int w  = X >> 3, fx = X & 7;
    const int h  = Y >> 3, fy = Y & 7;

    const float iters = (float)itersPtr[0];
    const float inv_temp = 1.0f / (1.0f + __expf(-(iters - 1.0f)));

    // mask channel index: k*64 + fy*8 + fx (k = 3x3 neighbor, row-major dy,dx)
    const float* __restrict__ mb =
        mask + (size_t)b * (9 * FAC * FAC) * HW + (size_t)(fy * FAC + fx) * HW
             + (size_t)h * Wl + w;

    float mv[9];
    float mmax = -3.4e38f;
    #pragma unroll
    for (int k = 0; k < 9; ++k) {
        const float v = mb[(size_t)k * (FAC * FAC) * HW] * inv_temp;
        mv[k] = v;
        mmax = fmaxf(mmax, v);
    }
    float denom = 0.0f;
    #pragma unroll
    for (int k = 0; k < 9; ++k) {
        mv[k] = __expf(mv[k] - mmax);
        denom += mv[k];
    }

    const float* __restrict__ db = disp + (size_t)b * HW;
    float accum = 0.0f;
    #pragma unroll
    for (int k = 0; k < 9; ++k) {
        const int dy = k / 3, dx = k % 3;
        const int hh = h + dy - 1;
        const int ww = w + dx - 1;
        float f = 0.0f;
        if (hh >= 0 && hh < Hh && ww >= 0 && ww < Wl)
            f = (float)FAC * db[(size_t)hh * Wl + ww];
        accum += mv[k] * f;
    }
    out[idx] = -(accum / denom);
}

// ---------------------------------------------------------------------------
extern "C" void kernel_launch(void* const* d_in, const int* in_sizes, int n_in,
                              void* d_out, int out_size, void* d_ws, size_t ws_size,
                              hipStream_t stream)
{
    const float* d_left  = (const float*)d_in[0];   // [B,C,H,W]
    const float* d_right = (const float*)d_in[1];   // [B,C,H,W]
    const float* d_mask  = (const float*)d_in[2];   // [B,576,H,W]
    const int*   d_iters = (const int*)  d_in[3];   // scalar

    float* d_cost = (float*)d_ws;                        // B*DISP*H*W floats (9.8 MB)
    float* d_disp = d_cost + (size_t)Bn * DISP * HW;     // B*H*W floats (0.4 MB)
    float* d_o    = (float*)d_out;                       // [B,1,8H,8W]

    // Stage 1: cost volume (one wave per (b,h,wtile))
    {
        dim3 grid(Bn * Hh * WT);
        dim3 block(32);
        cost_volume_wmma<<<grid, block, 0, stream>>>(d_left, d_right, d_cost);
    }
    // Stage 2: top-k regression
    {
        const int n = Bn * HW;
        topk_regress<<<(n + 255) / 256, 256, 0, stream>>>(d_cost, d_disp);
    }
    // Stage 3: convex upsample + negate
    {
        const int n = Bn * (FAC * Hh) * (FAC * Wl);
        convex_upsample_neg<<<(n + 255) / 256, 256, 0, stream>>>(d_disp, d_mask, d_iters, d_o);
    }
}
